// GraphCTHNodeV6Cell_58926951301239
// MI455X (gfx1250) — compile-verified, matrix operand order checked by reading, not run on hardware
//
#include <hip/hip_runtime.h>
#include <hip/hip_bf16.h>

// ---------------------------------------------------------------------------
// GraphCTHNodeV6Cell on MI455X (gfx1250, wave32, WMMA bf16).
//
// Strategy:
//  * Anorm converted once to bf16 (32MB -> L2-resident in 192MB L2).
//  * Per step t:  X1 = Anorm @ msg_in ; X2 = 2*Anorm @ X1 - msg_in
//    via v_wmma_f32_16x16x32_bf16 GEMMs (f32 accumulate).
//  * Fused per-16-node kernel: 4-path ChebConv message (K=480 WMMA),
//    softmax mix (LDS ds_add_f32), GRU gates (WMMA from LDS bf16), skip, pred.
// ---------------------------------------------------------------------------

typedef __attribute__((ext_vector_type(16))) __bf16 v16bf;
typedef __attribute__((ext_vector_type(8)))  __bf16 v8bf;
typedef __attribute__((ext_vector_type(8)))  float  v8f;

#define NN    4096
#define TT    64
#define HH    128
#define DINP  160   // DIN=131 padded to multiple of 32
#define KCAT  480   // 3 * DINP
#define G3    384   // 3*H
#define GKP   160   // GIN=132 padded

// --- helpers ---------------------------------------------------------------

__device__ __forceinline__ v16bf ldpair(const __bf16* lo, const __bf16* hi) {
  v8bf a = *(const v8bf*)lo;
  v8bf b = *(const v8bf*)hi;
  v16bf r;
#pragma unroll
  for (int i = 0; i < 8; ++i) { r[i] = a[i]; r[i + 8] = b[i]; }
  return r;
}

__device__ __forceinline__ v8f wmma_bf16(v16bf a, v16bf b, v8f c) {
  return __builtin_amdgcn_wmma_f32_16x16x32_bf16(
      false, a, false, b, (short)0, c, false, false);
}

__device__ __forceinline__ float sigmoidf_(float x) { return 1.f / (1.f + __expf(-x)); }

// --- setup kernels ---------------------------------------------------------

__global__ void k_rowsum(const float* __restrict__ A,
                         float* __restrict__ Dinv, float* __restrict__ degInv) {
  __shared__ float red[256];
  int n = blockIdx.x;
  float s = 0.f;
  for (int k = threadIdx.x; k < NN; k += 256) s += A[(size_t)n * NN + k];
  red[threadIdx.x] = s;
  __syncthreads();
  for (int w = 128; w > 0; w >>= 1) {
    if ((int)threadIdx.x < w) red[threadIdx.x] += red[threadIdx.x + w];
    __syncthreads();
  }
  if (threadIdx.x == 0) {
    float D = red[0];
    Dinv[n]   = (D > 0.f) ? 1.f / D : 0.f;
    degInv[n] = 1.f / (D + 1e-8f);
  }
}

__global__ void k_anorm(const float* __restrict__ A, const float* __restrict__ Dinv,
                        __bf16* __restrict__ Ab) {
  size_t i = (size_t)blockIdx.x * 256 + threadIdx.x;
  size_t stride = (size_t)gridDim.x * 256;
  for (; i < (size_t)NN * NN; i += stride)
    Ab[i] = (__bf16)(A[i] * Dinv[i >> 12]);
}

__global__ void k_mprop(const float* __restrict__ A, const float* __restrict__ m_seq,
                        const float* __restrict__ degInv, float* __restrict__ mprop) {
  __shared__ float red[256];
  int n = blockIdx.x;
  int t = threadIdx.x & 63, seg = threadIdx.x >> 6;
  float s = 0.f;
  for (int k = seg; k < NN; k += 4) s += A[(size_t)n * NN + k] * m_seq[(size_t)k * TT + t];
  red[threadIdx.x] = s;
  __syncthreads();
  if (seg == 0) {
    s = red[t] + red[t + 64] + red[t + 128] + red[t + 192];
    mprop[(size_t)n * TT + t] = s * degInv[n];
  }
}

// W_msg (4,3,131,128) -> WmsgT[p][h][k*160+d] bf16, zero padded
__global__ void k_pack_wmsg(const float* __restrict__ W, __bf16* __restrict__ Wt) {
  int idx = blockIdx.x * 256 + threadIdx.x;
  if (idx >= 4 * HH * KCAT) return;
  int p = idx / (HH * KCAT);
  int rem = idx % (HH * KCAT);
  int h = rem / KCAT, kk = rem % KCAT;
  int k = kk / DINP, d = kk % DINP;
  float v = (d < 131) ? W[(((size_t)p * 3 + k) * 131 + d) * HH + h] : 0.f;
  Wt[idx] = (__bf16)v;
}

// W_ih (384,132) -> WihT[r][k(160)] bf16 zero padded
__global__ void k_pack_wih(const float* __restrict__ W, __bf16* __restrict__ Wt) {
  int idx = blockIdx.x * 256 + threadIdx.x;
  if (idx >= G3 * GKP) return;
  int r = idx / GKP, k = idx % GKP;
  Wt[idx] = (__bf16)((k < 132) ? W[(size_t)r * 132 + k] : 0.f);
}

// W_hh (384,128) -> bf16
__global__ void k_pack_whh(const float* __restrict__ W, __bf16* __restrict__ Wt) {
  int idx = blockIdx.x * 256 + threadIdx.x;
  if (idx >= G3 * HH) return;
  Wt[idx] = (__bf16)W[idx];
}

// h0 = 0; msg_in(t=0) = [h0 | m0 tf0 tj0 | pad] in both layouts
__global__ void k_init(const float* __restrict__ m_seq, const float* __restrict__ tf,
                       const float* __restrict__ tj, float* __restrict__ hbuf,
                       __bf16* __restrict__ Xcat, __bf16* __restrict__ msginT) {
  int n = blockIdx.x * blockDim.x + threadIdx.x;
  if (n >= NN) return;
  for (int hh = 0; hh < HH; ++hh) hbuf[(size_t)n * HH + hh] = 0.f;
  __bf16* xr = Xcat + (size_t)n * KCAT;
  for (int d = 0; d < DINP; ++d) {
    float v = 0.f;
    if (d == 128) v = m_seq[(size_t)n * TT];
    else if (d == 129) v = tf[(size_t)n * TT];
    else if (d == 130) v = tj[(size_t)n * TT];
    __bf16 b = (__bf16)v;
    xr[d] = b;
    msginT[(size_t)d * NN + n] = b;
  }
}

// --- diffusion GEMM: out = Anorm(bf16,[4096x4096]) @ B  (B fed transposed) --
// stage 1: X1 -> Xcat[:,160:320] (row major) + X1T (transposed, B operand of stage 2)
// stage 2: X2 = 2*acc - X0 -> Xcat[:,320:480]
// grid (32,2) x 256thr: 8 waves/block, wave tile = 16 rows x 80 cols, K=4096.
__global__ void k_gemm(const __bf16* __restrict__ Anorm, const __bf16* __restrict__ Bt,
                       __bf16* __restrict__ Xcat, __bf16* __restrict__ X1T, int stage) {
  const int lane = threadIdx.x & 31;
  const int wave = threadIdx.x >> 5;
  const int m0 = (blockIdx.x * 8 + wave) * 16;
  const int d0 = blockIdx.y * 80;
  const int sel = lane >> 4;
  const int lr  = lane & 15;

  v8f acc[5];
#pragma unroll
  for (int j = 0; j < 5; ++j) acc[j] = (v8f)(0.f);

  const __bf16* arow = Anorm + (size_t)(m0 + lr) * NN + sel * 8;
  const __bf16* brow[5];
#pragma unroll
  for (int j = 0; j < 5; ++j)
    brow[j] = Bt + (size_t)(d0 + j * 16 + lr) * NN + sel * 8;

  for (int kk = 0; kk < NN; kk += 32) {
    v16bf af = ldpair(arow + kk, arow + kk + 16);
#pragma unroll
    for (int j = 0; j < 5; ++j) {
      v16bf bfv = ldpair(brow[j] + kk, brow[j] + kk + 16);
      acc[j] = wmma_bf16(af, bfv, acc[j]);
    }
  }

#pragma unroll
  for (int j = 0; j < 5; ++j) {
    int d = d0 + j * 16 + lr;
#pragma unroll
    for (int r = 0; r < 8; ++r) {
      int node = m0 + r + sel * 8;
      float v = acc[j][r];
      if (stage == 1) {
        __bf16 b = (__bf16)v;
        Xcat[(size_t)node * KCAT + DINP + d] = b;
        X1T[(size_t)d * NN + node] = b;
      } else {
        float x0 = (float)Xcat[(size_t)node * KCAT + d];
        Xcat[(size_t)node * KCAT + 2 * DINP + d] = (__bf16)(2.f * v - x0);
      }
    }
  }
}

// --- fused message + mix + GRU + skip + pred; 16 nodes / block (4 waves) ----
__global__ void __launch_bounds__(128)
k_msggru(int t, __bf16* __restrict__ Xcat, __bf16* __restrict__ msginT,
         float* __restrict__ hbuf, const float* __restrict__ mprop,
         const __bf16* __restrict__ WmsgT, const __bf16* __restrict__ WihT,
         const __bf16* __restrict__ WhhT,
         const float* __restrict__ x_seq, const float* __restrict__ m_seq,
         const float* __restrict__ tf_seq, const float* __restrict__ tj_seq,
         const float* __restrict__ b_msg, const float* __restrict__ path_bias,
         const float* __restrict__ mix_W, const float* __restrict__ mix_b,
         const float* __restrict__ b_ih, const float* __restrict__ b_hh,
         const float* __restrict__ out_W, const float* __restrict__ out_b,
         float* __restrict__ pred_out) {
  __shared__ __align__(16) float  s_logw[4][16];
  __shared__ __align__(16) float  s_msg[16][HH];
  __shared__ __align__(16) __bf16 s_inp[16][GKP];
  __shared__ __align__(16) __bf16 s_h[16][HH];
  __shared__ __align__(16) float  s_grz[16][256];
  __shared__ __align__(16) float  s_in[16][HH];
  __shared__ __align__(16) float  s_hn[16][HH];
  __shared__ float s_pred[16];

  const int tid  = threadIdx.x;
  const int lane = tid & 31;
  const int wave = tid >> 5;        // wave == path p in phase 1
  const int sel  = lane >> 4;
  const int lr   = lane & 15;
  const int node0 = blockIdx.x * 16;

  // zero accumulation LDS
  for (int i = tid; i < 16 * HH; i += 128) s_msg[i >> 7][i & 127] = 0.f;
  if (tid < 16) s_pred[tid] = 0.f;

  // ---- phase 1: m4[p] = tanh(Xcat[16x480] @ WmsgT[p]^T + b_msg[p]) --------
  v8f acc[8];
#pragma unroll
  for (int j = 0; j < 8; ++j) acc[j] = (v8f)(0.f);

  const __bf16* xrow = Xcat + (size_t)(node0 + lr) * KCAT + sel * 8;
  for (int kk = 0; kk < KCAT; kk += 32) {
    v16bf af = ldpair(xrow + kk, xrow + kk + 16);
#pragma unroll
    for (int j = 0; j < 8; ++j) {
      const __bf16* wrow =
          WmsgT + ((size_t)wave * HH + j * 16 + lr) * KCAT + sel * 8;
      v16bf bfv = ldpair(wrow + kk, wrow + kk + 16);
      acc[j] = wmma_bf16(af, bfv, acc[j]);
    }
  }
#pragma unroll
  for (int j = 0; j < 8; ++j) {
    float bm = b_msg[wave * HH + j * 16 + lr];
#pragma unroll
    for (int r = 0; r < 8; ++r) acc[j][r] = tanhf(acc[j][r] + bm);
  }

  // ---- phase 2: logits + softmax mix weights ------------------------------
  if (lane < 16) {
    const float* hrow = hbuf + (size_t)(node0 + lane) * HH;
    const float* mw = mix_W + wave * HH;
    float s = 0.f;
    for (int hh = 0; hh < HH; ++hh) s += hrow[hh] * mw[hh];
    s_logw[wave][lane] = s + mix_b[wave] + path_bias[wave];
  }
  __syncthreads();
  if (tid < 16) {
    float l0 = s_logw[0][tid], l1 = s_logw[1][tid];
    float l2 = s_logw[2][tid], l3 = s_logw[3][tid];
    float mx = fmaxf(fmaxf(l0, l1), fmaxf(l2, l3));
    float e0 = __expf(l0 - mx), e1 = __expf(l1 - mx);
    float e2 = __expf(l2 - mx), e3 = __expf(l3 - mx);
    float inv = 1.f / (e0 + e1 + e2 + e3);
    s_logw[0][tid] = e0 * inv; s_logw[1][tid] = e1 * inv;
    s_logw[2][tid] = e2 * inv; s_logw[3][tid] = e3 * inv;
  }
  __syncthreads();

  // ---- phase 3: msg = sum_p w_p * m4_p  (ds_add_f32 reduction) ------------
  float w8[8];
#pragma unroll
  for (int r = 0; r < 8; ++r) w8[r] = s_logw[wave][r + sel * 8];
#pragma unroll
  for (int j = 0; j < 8; ++j) {
    int hh = j * 16 + lr;
#pragma unroll
    for (int r = 0; r < 8; ++r)
      atomicAdd(&s_msg[r + sel * 8][hh], acc[j][r] * w8[r]);
  }
  __syncthreads();

  // ---- phase 4: stage bf16 GRU A-operands ---------------------------------
  for (int i = tid; i < 16 * GKP; i += 128) {
    int node = i / GKP, d = i % GKP;
    int gn = node0 + node;
    float v;
    if (d < 128)       v = s_msg[node][d];
    else if (d == 128) v = x_seq[(size_t)gn * TT + t];
    else if (d == 129) v = m_seq[(size_t)gn * TT + t];
    else if (d == 130) v = tf_seq[(size_t)gn * TT + t];
    else if (d == 131) v = tj_seq[(size_t)gn * TT + t];
    else               v = 0.f;
    s_inp[node][d] = (__bf16)v;
  }
  for (int i = tid; i < 16 * HH; i += 128) {
    int node = i >> 7, hh = i & 127;
    s_h[node][hh] = (__bf16)hbuf[(size_t)(node0 + node) * HH + hh];
  }
  __syncthreads();

  // ---- phase 5: GRU gates, wave w owns output cols [w*96, w*96+96) --------
  v8f gi[6], gh[6];
#pragma unroll
  for (int q = 0; q < 6; ++q) { gi[q] = (v8f)(0.f); gh[q] = (v8f)(0.f); }

  const __bf16* inrow = &s_inp[lr][0] + sel * 8;
  for (int kk = 0; kk < GKP; kk += 32) {
    v16bf af = ldpair(inrow + kk, inrow + kk + 16);
#pragma unroll
    for (int q = 0; q < 6; ++q) {
      int col = (wave * 6 + q) * 16 + lr;
      const __bf16* brow = WihT + (size_t)col * GKP + sel * 8;
      v16bf bfv = ldpair(brow + kk, brow + kk + 16);
      gi[q] = wmma_bf16(af, bfv, gi[q]);
    }
  }
  const __bf16* hrow16 = &s_h[lr][0] + sel * 8;
  for (int kk = 0; kk < HH; kk += 32) {
    v16bf af = ldpair(hrow16 + kk, hrow16 + kk + 16);
#pragma unroll
    for (int q = 0; q < 6; ++q) {
      int col = (wave * 6 + q) * 16 + lr;
      const __bf16* brow = WhhT + (size_t)col * HH + sel * 8;
      v16bf bfv = ldpair(brow + kk, brow + kk + 16);
      gh[q] = wmma_bf16(af, bfv, gh[q]);
    }
  }
#pragma unroll
  for (int q = 0; q < 6; ++q) {
    int col = (wave * 6 + q) * 16 + lr;
    float bi = b_ih[col], bh = b_hh[col];
#pragma unroll
    for (int r = 0; r < 8; ++r) {
      int node = r + sel * 8;
      float a = gi[q][r] + bi, b = gh[q][r] + bh;
      if (col < 256)      s_grz[node][col] = a + b;
      else              { s_in[node][col - 256] = a; s_hn[node][col - 256] = b; }
    }
  }
  __syncthreads();

  // ---- phase 6: gates, state update, skip, pred, next msg_in --------------
  for (int i = tid; i < 16 * HH; i += 128) {
    int node = i >> 7, hh = i & 127;
    int gn = node0 + node;
    float r  = sigmoidf_(s_grz[node][hh]);
    float z  = sigmoidf_(s_grz[node][hh + 128]);
    float nc = tanhf(s_in[node][hh] + r * s_hn[node][hh]);
    float hp = hbuf[(size_t)gn * HH + hh];
    float mt = m_seq[(size_t)gn * TT + t];
    float mp = mprop[(size_t)gn * TT + t];
    float skip = (0.1f + 0.05f * (1.f - mt)) * (1.f - 0.3f * mp);
    float h2 = (1.f - z) * nc + z * hp + skip * hp;
    hbuf[(size_t)gn * HH + hh] = h2;
    atomicAdd(&s_pred[node], h2 * out_W[hh]);
    if (t + 1 < TT) {
      __bf16 hb = (__bf16)h2;
      Xcat[(size_t)gn * KCAT + hh] = hb;
      msginT[(size_t)hh * NN + gn] = hb;
    }
  }
  if (t + 1 < TT && tid < 16) {
    int gn = node0 + tid;
    float m1 = m_seq[(size_t)gn * TT + t + 1];
    float f1 = tf_seq[(size_t)gn * TT + t + 1];
    float j1 = tj_seq[(size_t)gn * TT + t + 1];
    __bf16* xr = Xcat + (size_t)gn * KCAT;
    xr[128] = (__bf16)m1; xr[129] = (__bf16)f1; xr[130] = (__bf16)j1;
    msginT[(size_t)128 * NN + gn] = (__bf16)m1;
    msginT[(size_t)129 * NN + gn] = (__bf16)f1;
    msginT[(size_t)130 * NN + gn] = (__bf16)j1;
    for (int d = 131; d < DINP; ++d) {
      xr[d] = (__bf16)0.f;
      msginT[(size_t)d * NN + gn] = (__bf16)0.f;
    }
  }
  __syncthreads();
  if (tid < 16)
    pred_out[(size_t)(node0 + tid) * TT + t] = s_pred[tid] + out_b[0];
}

// ---------------------------------------------------------------------------

extern "C" void kernel_launch(void* const* d_in, const int* in_sizes, int n_in,
                              void* d_out, int out_size, void* d_ws, size_t ws_size,
                              hipStream_t stream) {
  const float* x_seq  = (const float*)d_in[0];
  const float* m_seq  = (const float*)d_in[1];
  const float* tf_seq = (const float*)d_in[2];
  const float* tj_seq = (const float*)d_in[3];
  const float* A      = (const float*)d_in[4];
  const float* W_msg  = (const float*)d_in[5];
  const float* b_msg  = (const float*)d_in[6];
  const float* pbias  = (const float*)d_in[7];
  const float* mix_W  = (const float*)d_in[8];
  const float* mix_b  = (const float*)d_in[9];
  const float* W_ih   = (const float*)d_in[10];
  const float* W_hh   = (const float*)d_in[11];
  const float* b_ih   = (const float*)d_in[12];
  const float* b_hh   = (const float*)d_in[13];
  const float* out_W  = (const float*)d_in[14];
  const float* out_b  = (const float*)d_in[15];

  // workspace carve (~44 MB total)
  char* ws = (char*)d_ws;
  auto carve = [&](size_t bytes) -> char* {
    char* p = ws;
    ws += (bytes + 255) & ~(size_t)255;
    return p;
  };
  __bf16* Anorm  = (__bf16*)carve((size_t)NN * NN * 2);
  __bf16* Xcat   = (__bf16*)carve((size_t)NN * KCAT * 2);
  __bf16* msginT = (__bf16*)carve((size_t)DINP * NN * 2);
  __bf16* X1T    = (__bf16*)carve((size_t)DINP * NN * 2);
  float*  hbuf   = (float*)carve((size_t)NN * HH * 4);
  float*  mprop  = (float*)carve((size_t)NN * TT * 4);
  float*  Dinv   = (float*)carve((size_t)NN * 4);
  float*  degInv = (float*)carve((size_t)NN * 4);
  __bf16* WmsgT  = (__bf16*)carve((size_t)4 * HH * KCAT * 2);
  __bf16* WihT   = (__bf16*)carve((size_t)G3 * GKP * 2);
  __bf16* WhhT   = (__bf16*)carve((size_t)G3 * HH * 2);

  k_rowsum<<<NN, 256, 0, stream>>>(A, Dinv, degInv);
  k_anorm<<<8192, 256, 0, stream>>>(A, Dinv, Anorm);
  k_mprop<<<NN, 256, 0, stream>>>(A, m_seq, degInv, mprop);
  k_pack_wmsg<<<(4 * HH * KCAT + 255) / 256, 256, 0, stream>>>(W_msg, WmsgT);
  k_pack_wih<<<(G3 * GKP + 255) / 256, 256, 0, stream>>>(W_ih, WihT);
  k_pack_whh<<<(G3 * HH + 255) / 256, 256, 0, stream>>>(W_hh, WhhT);
  k_init<<<16, 256, 0, stream>>>(m_seq, tf_seq, tj_seq, hbuf, Xcat, msginT);

  for (int t = 0; t < TT; ++t) {
    k_gemm<<<dim3(32, 2), 256, 0, stream>>>(Anorm, msginT, Xcat, X1T, 1);
    k_gemm<<<dim3(32, 2), 256, 0, stream>>>(Anorm, X1T, Xcat, X1T, 2);
    k_msggru<<<NN / 16, 128, 0, stream>>>(
        t, Xcat, msginT, hbuf, mprop, WmsgT, WihT, WhhT,
        x_seq, m_seq, tf_seq, tj_seq, b_msg, pbias, mix_W, mix_b,
        b_ih, b_hh, out_W, out_b, (float*)d_out);
  }
}
